// SpikingModel_30915174597101
// MI455X (gfx1250) — matrix operand order repaired and checked by reading, not compiled, not run on hardware
//
#include <hip/hip_runtime.h>

// ---------------------------------------------------------------------------
// EIF spiking network, 500 sequential steps of Z = Y*W^T + x0 + EIF update.
// Strategy (MI455X / gfx1250):
//  * W converted once to bf16 (32 MB) -> stays resident in the 192 MB L2.
//  * One persistent kernel runs all 500 steps (device-wide barrier per step)
//    to avoid 500 kernel-launch latencies on a ~1us/step problem.
//  * GEMM done with v_wmma_f32_16x16x32_bf16, 256 j-tiles x 4 k-slices
//    over 128 blocks; per-block LDS staging of Y (bf16) + LDS split-K reduce.
//  * Neuron state V/Y/r lives in f32 registers of the owning wave.
// ---------------------------------------------------------------------------

typedef __attribute__((ext_vector_type(16))) __bf16       v16bf;
typedef __attribute__((ext_vector_type(8)))  __bf16       v8bf;
typedef __attribute__((ext_vector_type(8)))  float        v8f;
typedef __attribute__((ext_vector_type(4)))  unsigned int v4u;

namespace {
constexpr int   kN        = 4096;
constexpr int   kB        = 16;
constexpr int   kNT       = 500;
constexpr float kDt       = 0.1f;
constexpr float kTauSyn   = 8.0f;
constexpr float kTauM     = 10.0f;
constexpr float kEL       = -72.0f;
constexpr float kVth      = 0.0f;
constexpr float kVre      = -72.0f;
constexpr float kVT       = -55.0f;
constexpr float kDtEif    = 1.0f;
constexpr float kVlb      = -85.0f;
constexpr float kInvDt    = 10.0f;               // 1/dt
constexpr float kOutScale = 0.002f;              // dt / (NT*dt)

constexpr int kNBlocks       = 128;              // 2 j-tiles per block
constexpr int kWavesPerBlock = 8;                // 2 tiles x 4 k-slices
constexpr int kKSlice        = kN / 4;           // 1024 K per wave
constexpr int kRowStride     = kN + 16;          // shorts; 8224 B row, 32B-aligned
constexpr int kStageBytes    = kB * kRowStride * 2;          // 131584
constexpr int kRedBytes      = kWavesPerBlock * 32 * 8 * 4;  // 8192
constexpr int kLdsBytes      = kStageBytes + kRedBytes;
}

__device__ __forceinline__ unsigned short f32_to_bf16(float f) {
  unsigned u = __builtin_bit_cast(unsigned, f);
  u += 0x7FFFu + ((u >> 16) & 1u);               // round-to-nearest-even
  return (unsigned short)(u >> 16);
}

// Device-wide barrier: one single-use counter per step (re-zeroed by prep
// kernel every launch => deterministic, graph-replay safe, no sense reversal).
__device__ __forceinline__ void grid_barrier(unsigned int* bar, int slot) {
  __syncthreads();
  if (threadIdx.x == 0) {
    __threadfence();
    unsigned prev = atomicAdd(&bar[slot], 1u);
    if (prev + 1u < (unsigned)kNBlocks) {
      while (__hip_atomic_load(&bar[slot], __ATOMIC_RELAXED,
                               __HIP_MEMORY_SCOPE_AGENT) < (unsigned)kNBlocks) {
        __builtin_amdgcn_s_sleep(1);
      }
    }
    __threadfence();
  }
  __syncthreads();
}

// ---- one-time per launch: W f32 -> bf16, zero barrier counters -------------
__global__ __launch_bounds__(256) void prep_convert(
    const float* __restrict__ W, unsigned short* __restrict__ Wbf,
    unsigned int* __restrict__ bar) {
  if (blockIdx.x == 0) {
    bar[threadIdx.x] = 0u;
    bar[threadIdx.x + 256] = 0u;
  }
  size_t base = ((size_t)blockIdx.x * 256 + threadIdx.x) * 8;
  v4u p;
#pragma unroll
  for (int i = 0; i < 4; ++i) {
    unsigned lo = f32_to_bf16(W[base + 2 * i]);
    unsigned hi = f32_to_bf16(W[base + 2 * i + 1]);
    p[i] = lo | (hi << 16);
  }
  *(v4u*)(Wbf + base) = p;
}

// ---- persistent simulation kernel ------------------------------------------
__global__ __launch_bounds__(256, 1) void eif_persistent(
    const unsigned short* __restrict__ Wbf,
    const float* __restrict__ x0, const float* __restrict__ V0,
    const float* __restrict__ Y0, unsigned short* __restrict__ ybufA,
    unsigned short* __restrict__ ybufB, unsigned int* __restrict__ bar,
    float* __restrict__ out) {
  extern __shared__ char smem[];
  unsigned short* ldsY = (unsigned short*)smem;          // staged Y (bf16)
  float* red = (float*)(smem + kStageBytes);             // split-K partials

  const int tid    = threadIdx.x;
  const int lane   = tid & 31;
  const int wave   = tid >> 5;
  const int tileIB = wave >> 2;                          // 0..1
  const int kslice = wave & 3;                           // 0..3
  const int jt     = blockIdx.x * 2 + tileIB;            // 0..255 j-tile
  const int hi     = lane >> 4;
  const int jcol   = (jt << 4) + (lane & 15);
  const bool owner = (kslice == 0);                      // owns EIF state

  // Per-lane owned elements (C/D layout: M = v + 8*hi, N = lane%16).
  float V[8], r[8], Y[8], X[8];
  int idx[8];
  if (owner) {
#pragma unroll
    for (int v = 0; v < 8; ++v) {
      idx[v] = (v + 8 * hi) * kN + jcol;
      V[v] = V0[idx[v]];
      Y[v] = Y0[idx[v]];
      X[v] = x0[idx[v]];
      r[v] = 0.0f;
      ybufA[idx[v]] = f32_to_bf16(Y[v]);                 // publish Y(t=0)
    }
  }
  grid_barrier(bar, 0);

  // B fragment: lane holds column j, 16 contiguous K values (hi selects K half)
  const unsigned short* wbase =
      Wbf + (size_t)jcol * kN + (kslice * kKSlice + hi * 16);
  // A fragment: lane holds row M=lane%16; two contiguous 16B K-chunks
  const unsigned short* abase =
      ldsY + (lane & 15) * kRowStride + kslice * kKSlice;

  for (int t = 0; t < kNT; ++t) {
    const unsigned short* yin  = (t & 1) ? ybufB : ybufA;
    unsigned short*       yout = (t & 1) ? ybufA : ybufB;

    // Cooperative stage of Y (16 x 4096 bf16) into padded LDS.
#pragma unroll 4
    for (int i = 0; i < 32; ++i) {
      int c = tid + (i << 8);                            // 0..8191 16B chunks
      int row = c >> 9, col8 = c & 511;
      v4u d = *(const v4u*)(yin + row * kN + col8 * 8);
      *(v4u*)(ldsY + row * kRowStride + col8 * 8) = d;
    }
    __syncthreads();

    v8f acc = {0.f, 0.f, 0.f, 0.f, 0.f, 0.f, 0.f, 0.f};
#pragma unroll 4
    for (int kc = 0; kc < kKSlice / 32; ++kc) {
      v8bf alo = *(const v8bf*)(abase + kc * 32 + hi * 16);
      v8bf ahi = *(const v8bf*)(abase + kc * 32 + 8 + hi * 16);
      v16bf a = __builtin_shufflevector(alo, ahi, 0, 1, 2, 3, 4, 5, 6, 7, 8, 9,
                                        10, 11, 12, 13, 14, 15);
      v16bf b = *(const v16bf*)(wbase + kc * 32);
      acc = __builtin_amdgcn_wmma_f32_16x16x32_bf16(false, a, false, b,
                                                    (short)0, acc, false, false);
    }

    // Split-K reduction through LDS.
    float* myred = red + (wave * 32 + lane) * 8;
#pragma unroll
    for (int v = 0; v < 8; ++v) myred[v] = acc[v];
    __syncthreads();

    if (owner) {
#pragma unroll
      for (int v = 0; v < 8; ++v) {
        const int rb = ((tileIB * 4) * 32 + lane) * 8 + v;
        float Z = red[rb] + red[rb + 256] + red[rb + 512] + red[rb + 768] + X[v];
        float vv = V[v];
        float dV = (kEL - vv + kDtEif * __expf((vv - kVT) * (1.0f / kDtEif)) + Z)
                   * (1.0f / kTauM);
        vv = fmaxf(vv + kDt * dV, kVlb);
        const bool spike = vv >= kVth;
        const float S = spike ? kInvDt : 0.0f;
        V[v] = spike ? kVre : vv;
        r[v] += S;
        Y[v] += (kDt / kTauSyn) * (S - Y[v]);
        yout[idx[v]] = f32_to_bf16(Y[v]);                // publish Y(t+1)
      }
    }
    grid_barrier(bar, t + 1);                            // Y(t+1) visible
  }

  if (owner) {
#pragma unroll
    for (int v = 0; v < 8; ++v) out[idx[v]] = r[v] * kOutScale;
  }
}

extern "C" void kernel_launch(void* const* d_in, const int* in_sizes, int n_in,
                              void* d_out, int out_size, void* d_ws,
                              size_t ws_size, hipStream_t stream) {
  (void)in_sizes; (void)n_in; (void)out_size; (void)ws_size;
  const float* W  = (const float*)d_in[0];
  const float* x0 = (const float*)d_in[1];
  const float* V0 = (const float*)d_in[2];
  const float* Y0 = (const float*)d_in[3];
  float* out = (float*)d_out;

  char* ws = (char*)d_ws;
  unsigned short* Wbf = (unsigned short*)ws;                       // 32 MB
  unsigned short* ybA = (unsigned short*)(ws + (size_t)32 * 1024 * 1024);
  unsigned short* ybB = (unsigned short*)(ws + (size_t)32 * 1024 * 1024 + 131072);
  unsigned int*   bar = (unsigned int*)(ws + (size_t)32 * 1024 * 1024 + 262144);

  prep_convert<<<(kN * kN) / (256 * 8), 256, 0, stream>>>(W, Wbf, bar);
  eif_persistent<<<kNBlocks, 256, kLdsBytes, stream>>>(Wbf, x0, V0, Y0, ybA,
                                                       ybB, bar, out);
}
